// RNN_36593121362136
// MI455X (gfx1250) — matrix operand order repaired and checked by reading, not compiled, not run on hardware
//
#include <hip/hip_runtime.h>
#include <hip/hip_bf16.h>

// ---------------------------------------------------------------------------
// Problem dims (from reference): SEQ=512, B=64, IN=256, H=512, L=2
// ---------------------------------------------------------------------------
#define ST   512   // timesteps
#define SB   64    // batch
#define SIN  256   // input features
#define SH   512   // hidden
#define NSLICE 32  // N columns per scan workgroup
#define NBLK  (SH / NSLICE)   // 16 scan workgroups

typedef __bf16 bf16;
typedef __attribute__((ext_vector_type(16))) __bf16 v16bf;
typedef __attribute__((ext_vector_type(8)))  float  v8f;

union frag_u {
    v16bf v;
    uint4 q[2];
};

// ---------------------------------------------------------------------------
// f32 -> bf16 (round-to-nearest-even) and exact GELU
// ---------------------------------------------------------------------------
__device__ __forceinline__ bf16 f2bf(float x) {
    unsigned u = __builtin_bit_cast(unsigned, x);
    unsigned r = u + 0x7fffu + ((u >> 16) & 1u);
    unsigned short h = (unsigned short)(r >> 16);
    return __builtin_bit_cast(bf16, h);
}

__device__ __forceinline__ float gelu_exact(float v) {
    return 0.5f * v * (1.0f + erff(v * 0.70710678118654752440f));
}

// ---------------------------------------------------------------------------
// WMMA fragment loaders, vectorized per CDNA5 ISA §7.12.2 (wave32).
//
// A (16x32, row-major LDS, row stride ld elems, 16B-aligned rows):
//   lane l: row m=l&15, half h=l>>4 holds K runs [8h,8h+8) and [16+8h,16+8h+8)
//   -> two ds_load_b128.
// B (32x16) from a TRANSPOSED [N][K] tile (row stride ldT elems):
//   lane l: column n=l&15, half h=l>>4 holds K runs [8h,8h+8) and [16+8h,16+8h+8)
//   -> two ds_load_b128.
// ---------------------------------------------------------------------------
__device__ __forceinline__ v16bf load_a_frag(const bf16* base, int ld, int lane) {
    const int m    = lane & 15;
    const int half = lane >> 4;
    const bf16* p  = base + m * ld + 8 * half;
    frag_u f;
    f.q[0] = *(const uint4*)p;          // K = 8h .. 8h+7
    f.q[1] = *(const uint4*)(p + 16);   // K = 16+8h .. 16+8h+7
    return f.v;
}

__device__ __forceinline__ v16bf load_b_fragT(const bf16* baseT, int ldT, int lane) {
    const int n  = lane & 15;
    const int kh = (lane >> 4) * 8;
    const bf16* p = baseT + n * ldT + kh;
    frag_u f;
    f.q[0] = *(const uint4*)p;          // K = kh .. kh+7
    f.q[1] = *(const uint4*)(p + 16);   // K = 16+kh .. 16+kh+7
    return f.v;
}

#define WMMA_BF16(A, B, C) \
    __builtin_amdgcn_wmma_f32_16x16x32_bf16(false, (A), false, (B), (short)0, (C), false, false)

// ---------------------------------------------------------------------------
// Converters
// ---------------------------------------------------------------------------
__global__ void k_f32_to_bf16(const float* __restrict__ src,
                              bf16* __restrict__ dst, int n) {
    int i = blockIdx.x * blockDim.x + threadIdx.x;
    if (i < n) dst[i] = f2bf(src[i]);
}

// src [R][C] f32 (row-major) -> dst [C][R] bf16 (transposed, row-major)
__global__ void k_f32_to_bf16_T(const float* __restrict__ src,
                                bf16* __restrict__ dst, int R, int C) {
    int i = blockIdx.x * blockDim.x + threadIdx.x;
    if (i < R * C) {
        int r = i / C, c = i % C;
        dst[(size_t)c * R + r] = f2bf(src[i]);
    }
}

__global__ void k_init_bar(unsigned* p) {
    if (threadIdx.x < 2) p[threadIdx.x] = 0u;
}

// ---------------------------------------------------------------------------
// GEMM: O[M,N] (f32) = X[M,K] (bf16, row-major) @ W (bf16, PRE-TRANSPOSED
// Wt[N][K] row-major). 256 threads (8 waves), 64x64 tile, K-chunks of 32,
// DOUBLE-BUFFERED LDS: prefetch chunk k+1 into registers while chunk k feeds
// the WMMA chain; one workgroup barrier per chunk.
// wave w: M-tile = (w>>1)*16, N-tiles = (w&1)*32 and (w&1)*32+16.
// ---------------------------------------------------------------------------
__global__ void k_gemm_xw(const bf16* __restrict__ X, const bf16* __restrict__ Wt,
                          float* __restrict__ O, int M, int N, int K) {
    __shared__ __align__(16) bf16 As [2][64 * 32];   // [m][k]
    __shared__ __align__(16) bf16 Bst[2][64 * 32];   // [n][k] (transposed tile)

    const int tid  = threadIdx.x;
    const int lane = tid & 31;
    const int wave = tid >> 5;
    const int m0   = blockIdx.x * 64;
    const int n0   = blockIdx.y * 64;

    const int wtm = (wave >> 1) * 16;  // 0,16,32,48
    const int wtn = (wave & 1) * 32;   // 0 or 32

    const int r8 = tid >> 2;           // 0..63
    const int c8 = (tid & 3) * 8;      // 0,8,16,24

    const bf16* pX = X  + (size_t)(m0 + r8) * K + c8;
    const bf16* pW = Wt + (size_t)(n0 + r8) * K + c8;

    v8f acc0 = {};
    v8f acc1 = {};

    // Stage chunk 0 into buffer 0.
    *(uint4*)&As [0][r8 * 32 + c8] = *(const uint4*)pX;
    *(uint4*)&Bst[0][r8 * 32 + c8] = *(const uint4*)pW;
    __syncthreads();

    for (int k0 = 0; k0 < K; k0 += 32) {
        const int  buf  = (k0 >> 5) & 1;
        const bool more = (k0 + 32) < K;

        // Prefetch next chunk into registers (overlaps with WMMA below).
        uint4 na, nb;
        if (more) {
            na = *(const uint4*)(pX + k0 + 32);
            nb = *(const uint4*)(pW + k0 + 32);
        }

        v16bf a  = load_a_frag (&As [buf][wtm * 32], 32, lane);
        v16bf b0 = load_b_fragT(&Bst[buf][wtn * 32], 32, lane);
        v16bf b1 = load_b_fragT(&Bst[buf][(wtn + 16) * 32], 32, lane);
        acc0 = WMMA_BF16(a, b0, acc0);
        acc1 = WMMA_BF16(a, b1, acc1);

        if (more) {
            // Previous iteration's trailing barrier guarantees all reads of
            // buf^1 have drained; safe to overwrite, then publish.
            *(uint4*)&As [buf ^ 1][r8 * 32 + c8] = na;
            *(uint4*)&Bst[buf ^ 1][r8 * 32 + c8] = nb;
            __syncthreads();
        }
    }

    const int cm  = lane & 15;
    const int chf = lane >> 4;
#pragma unroll
    for (int r = 0; r < 8; ++r) {
        int gm = m0 + wtm + r + 8 * chf;
        O[(size_t)gm * N + n0 + wtn + cm]      = acc0[r];
        O[(size_t)gm * N + n0 + wtn + 16 + cm] = acc1[r];
    }
}

// ---------------------------------------------------------------------------
// Global (grid-wide) generation barrier for the persistent scan kernel.
// ---------------------------------------------------------------------------
__device__ __forceinline__ void grid_barrier(unsigned* cnt, unsigned* gen, unsigned nblk) {
    __syncthreads();
    if (threadIdx.x == 0) {
        __threadfence();
        unsigned g = __hip_atomic_load(gen, __ATOMIC_RELAXED, __HIP_MEMORY_SCOPE_AGENT);
        if (__hip_atomic_fetch_add(cnt, 1u, __ATOMIC_ACQ_REL, __HIP_MEMORY_SCOPE_AGENT) == nblk - 1u) {
            __hip_atomic_store(cnt, 0u, __ATOMIC_RELAXED, __HIP_MEMORY_SCOPE_AGENT);
            __hip_atomic_fetch_add(gen, 1u, __ATOMIC_RELEASE, __HIP_MEMORY_SCOPE_AGENT);
        } else {
            while (__hip_atomic_load(gen, __ATOMIC_ACQUIRE, __HIP_MEMORY_SCOPE_AGENT) == g)
                __builtin_amdgcn_s_sleep(2);
        }
        __threadfence();
    }
    __syncthreads();
}

// ---------------------------------------------------------------------------
// Persistent recurrent scan: h_{t+1} = gelu(h_t @ Wh + xin_t)
// Grid = 16 blocks x 256 threads (8 waves); block g owns columns [g*32, g*32+32).
// LDS: Wh slice TRANSPOSED [32][512] bf16 (32KB, one contiguous slab of Wh_t,
// loaded once) + full h [64][512] bf16 (64KB, restaged per step).
// Hidden state is DOUBLE-BUFFERED in global memory (hb0/hb1, alternating per
// step), so only ONE grid barrier per timestep is needed: by the time step
// t+1's barrier completes, every block has finished its step-t reads, so
// step t+2 may safely overwrite the step-t buffer.
// Each wave: one 16x16 C tile, 16 chained v_wmma_f32_16x16x32_bf16 per step.
// ---------------------------------------------------------------------------
__global__ void k_rnn_scan(const float* __restrict__ xin,    // [T,B,H] f32
                           const bf16* __restrict__ Wh_t,    // [H][H] bf16, TRANSPOSED (n-major)
                           const float* __restrict__ h0,     // [H] f32 (broadcast over batch)
                           bf16* __restrict__ hb0,           // [B,H] bf16 state buffer (even t)
                           bf16* __restrict__ hb1,           // [B,H] bf16 state buffer (odd t)
                           bf16* __restrict__ out_bf,        // [T,B,H] bf16 or nullptr
                           float* __restrict__ out_f,        // [T,B,H] f32 or nullptr
                           float* __restrict__ hN,           // [B,H] f32 final hidden
                           unsigned* bar_cnt, unsigned* bar_gen) {
    extern __shared__ __align__(16) bf16 smem[];
    bf16* w_s = smem;                  // [NSLICE][SH]  (n-major, K contiguous)
    bf16* h_s = smem + NSLICE * SH;    // [SB][SH]

    const int tid  = threadIdx.x;
    const int lane = tid & 31;
    const int wave = tid >> 5;
    const int c0   = blockIdx.x * NSLICE;
    const unsigned nblk = gridDim.x;

    // Wh_t rows [c0, c0+32) are one contiguous 32KB slab -> b128 copy into LDS.
    {
        const uint4* src = (const uint4*)(Wh_t + (size_t)c0 * SH);
        for (int i = tid; i < (NSLICE * SH) / 8; i += blockDim.x)
            ((uint4*)w_s)[i] = src[i];
    }
    // Initial hidden state: broadcast h0 over the batch.
    for (int k = tid; k < SH; k += blockDim.x) {
        bf16 v = f2bf(h0[k]);
        for (int m = 0; m < SB; ++m) h_s[m * SH + k] = v;
    }
    __syncthreads();

    const int wtm = (wave & 3) * 16;   // M tile: 0,16,32,48
    const int wtn = (wave >> 2) * 16;  // local N tile: 0 or 16
    const int cm  = lane & 15;
    const int chf = lane >> 4;

    for (int t = 0; t < ST; ++t) {
        v8f acc = {};
#pragma unroll 4
        for (int kc = 0; kc < SH; kc += 32) {
            v16bf a = load_a_frag (&h_s[wtm * SH + kc], SH, lane);
            v16bf b = load_b_fragT(&w_s[wtn * SH + kc], SH, lane);
            acc = WMMA_BF16(a, b, acc);
        }

        bf16* hb = (t & 1) ? hb1 : hb0;   // this step's publication buffer
        const size_t tbase = (size_t)t * SB * SH;
#pragma unroll
        for (int r = 0; r < 8; ++r) {
            int m  = wtm + r + 8 * chf;       // batch row 0..63
            int gn = c0 + wtn + cm;           // hidden col 0..511
            size_t idx = tbase + (size_t)m * SH + gn;
            float v = acc[r] + xin[idx];
            float g = gelu_exact(v);
            if (out_f)  out_f[idx]  = g;
            if (out_bf) out_bf[idx] = f2bf(g);
            hb[(size_t)m * SH + gn] = f2bf(g);
            if (t == ST - 1) hN[(size_t)m * SH + gn] = g;
        }

        if (t < ST - 1) {
            grid_barrier(bar_cnt, bar_gen, nblk);   // all slices of h_{t+1} visible
            // Restage full hidden state into LDS (64x512 bf16 = 4096 uint4).
            for (int i = tid; i < (SB * SH) / 8; i += blockDim.x)
                ((uint4*)h_s)[i] = ((const uint4*)hb)[i];
            // No second barrier: step t+1 writes the OTHER buffer, and by the
            // time step t+1's barrier completes, these reads have drained.
        }
    }
}

// ---------------------------------------------------------------------------
// Host launcher
// ---------------------------------------------------------------------------
extern "C" void kernel_launch(void* const* d_in, const int* in_sizes, int n_in,
                              void* d_out, int out_size, void* d_ws, size_t ws_size,
                              hipStream_t stream) {
    (void)in_sizes; (void)n_in; (void)out_size; (void)ws_size;

    const float* seq   = (const float*)d_in[0];  // [T,B,IN]
    const float* w_in0 = (const float*)d_in[1];  // [IN,H]
    const float* w_h0  = (const float*)d_in[2];  // [H,H]
    const float* h0_0  = (const float*)d_in[3];  // [1,H]
    const float* w_in1 = (const float*)d_in[4];  // [H,H]
    const float* w_h1  = (const float*)d_in[5];  // [H,H]
    const float* h0_1  = (const float*)d_in[6];  // [1,H]

    float* out1 = (float*)d_out;                       // [T,B,H]
    float* hn0  = out1 + (size_t)ST * SB * SH;         // [B,H]
    float* hn1  = hn0 + (size_t)SB * SH;               // [B,H]

    // Workspace carve-up (256B-aligned chunks).
    char* ws = (char*)d_ws;
    size_t off = 0;
    auto carve = [&](size_t bytes) -> void* {
        void* p = ws + off;
        off += (bytes + 255) & ~(size_t)255;
        return p;
    };
    unsigned* bar       = (unsigned*)carve(256);
    float*    xin       = (float*)carve((size_t)ST * SB * SH * sizeof(float));  // 64 MB
    bf16*     xbf_seq   = (bf16*)carve((size_t)ST * SB * SIN * sizeof(bf16));   // 16 MB
    bf16*     out0_bf   = (bf16*)carve((size_t)ST * SB * SH * sizeof(bf16));    // 32 MB
    bf16*     wt_in0    = (bf16*)carve((size_t)SIN * SH * sizeof(bf16));  // [H][IN]
    bf16*     wt_h0     = (bf16*)carve((size_t)SH * SH * sizeof(bf16));   // [H][H] transposed
    bf16*     wt_in1    = (bf16*)carve((size_t)SH * SH * sizeof(bf16));   // [H][H] transposed
    bf16*     wt_h1     = (bf16*)carve((size_t)SH * SH * sizeof(bf16));   // [H][H] transposed
    bf16*     hb0       = (bf16*)carve((size_t)SB * SH * sizeof(bf16));   // state ping
    bf16*     hb1       = (bf16*)carve((size_t)SB * SH * sizeof(bf16));   // state pong

    const size_t scan_lds = (size_t)(NSLICE * SH + SB * SH) * sizeof(bf16);     // 96 KB
    hipFuncSetAttribute((const void*)k_rnn_scan,
                        hipFuncAttributeMaxDynamicSharedMemorySize, (int)scan_lds);

    k_init_bar<<<1, 32, 0, stream>>>(bar);

    // Activations: straight f32->bf16. Weights: f32->bf16 with transpose
    // (so the WMMA B operand is n-major / K-contiguous everywhere).
    k_f32_to_bf16<<<(ST * SB * SIN + 255) / 256, 256, 0, stream>>>(
        seq, xbf_seq, ST * SB * SIN);
    k_f32_to_bf16_T<<<(SIN * SH + 255) / 256, 256, 0, stream>>>(w_in0, wt_in0, SIN, SH);
    k_f32_to_bf16_T<<<(SH * SH + 255) / 256, 256, 0, stream>>>(w_h0,  wt_h0,  SH,  SH);
    k_f32_to_bf16_T<<<(SH * SH + 255) / 256, 256, 0, stream>>>(w_in1, wt_in1, SH,  SH);
    k_f32_to_bf16_T<<<(SH * SH + 255) / 256, 256, 0, stream>>>(w_h1,  wt_h1,  SH,  SH);

    // Layer 0: input projection xin = seq @ w_in0   (M=32768, N=512, K=256)
    k_gemm_xw<<<dim3((ST * SB) / 64, SH / 64), 256, 0, stream>>>(
        xbf_seq, wt_in0, xin, ST * SB, SH, SIN);

    // Layer 0 recurrence -> out0 (bf16) + hn0
    k_rnn_scan<<<NBLK, 256, scan_lds, stream>>>(
        xin, wt_h0, h0_0, hb0, hb1, out0_bf, nullptr, hn0, bar, bar + 1);

    // Layer 1: input projection xin = out0 @ w_in1  (M=32768, N=512, K=512)
    k_gemm_xw<<<dim3((ST * SB) / 64, SH / 64), 256, 0, stream>>>(
        out0_bf, wt_in1, xin, ST * SB, SH, SH);

    // Layer 1 recurrence -> out1 (f32, final output) + hn1
    k_rnn_scan<<<NBLK, 256, scan_lds, stream>>>(
        xin, wt_h1, h0_1, hb0, hb1, nullptr, out1, hn1, bar, bar + 1);
}
